// MultiHeadAttention_50749333570104
// MI455X (gfx1250) — matrix operand order repaired and checked by reading, not compiled, not run on hardware
//
#include <hip/hip_runtime.h>
#include <hip/hip_bf16.h>
#include <math.h>

// ---- problem constants (match reference) ----
#define S_LEN 2048
#define EMB   1024
#define HEADS 16
#define HDIM  64
#define BATCH 2
#define ROWS  (BATCH * S_LEN)   // 4096
#define QKV3  (3 * EMB)         // 3072

typedef __attribute__((ext_vector_type(16))) __bf16 v16bf;
typedef __attribute__((ext_vector_type(8)))  __bf16 v8bf;
typedef __attribute__((ext_vector_type(8)))  float  v8f;

// ---------------- fragment loaders ----------------
// A-fragment (16x32 bf16, row-major source, leading dim `ld`):
// lane L (l15=L&15, hi=L>>4) holds row (row0+l15),
// K = kbase + hi*8 + {0..7}  and  K = kbase + 16 + hi*8 + {0..7}
static __device__ inline v16bf load_a_frag(const __bf16* base, int ld, int row0,
                                           int kbase, int l15, int hi) {
  const __bf16* p = base + (size_t)(row0 + l15) * ld + kbase + hi * 8;
  v8bf lo = *reinterpret_cast<const v8bf*>(p);
  v8bf hh = *reinterpret_cast<const v8bf*>(p + 16);
  v16bf r;
#pragma unroll
  for (int i = 0; i < 8; ++i) { r[i] = lo[i]; r[i + 8] = hh[i]; }
  return r;
}

// B-fragment (32x16 bf16) from N-major storage, leading dim `ld`:
// lane L holds column (n0+l15), K = kbase + hi*8 + {0..7} and K = kbase + 16 + hi*8 + {0..7}.
static __device__ inline v16bf load_b_frag(const __bf16* baseT, int ld, int n0,
                                           int kbase, int l15, int hi) {
  const __bf16* p = baseT + (size_t)(n0 + l15) * ld + kbase + hi * 8;
  v8bf lo = *reinterpret_cast<const v8bf*>(p);
  v8bf hh = *reinterpret_cast<const v8bf*>(p + 16);
  v16bf r;
#pragma unroll
  for (int i = 0; i < 8; ++i) { r[i] = lo[i]; r[i + 8] = hh[i]; }
  return r;
}

static __device__ inline v8f wmma_bf16(v16bf a, v16bf b, v8f c) {
  return __builtin_amdgcn_wmma_f32_16x16x32_bf16(false, a, false, b, (short)0, c,
                                                 false, false);
}

static __device__ inline v8f zero8() {
  v8f z;
#pragma unroll
  for (int i = 0; i < 8; ++i) z[i] = 0.0f;
  return z;
}

// ---------------- CDNA5 async global->LDS copy ----------------
// 16 bytes per lane, tracked on ASYNCcnt. LDS destination byte offset is the
// low 32 bits of the generic shared-memory pointer (flat->LDS aperture
// truncation, ISA 10.2).
static __device__ inline void async_copy_b128(const __bf16* g, const __bf16* l) {
  unsigned ldsoff = (unsigned)(uintptr_t)l;
  unsigned long long ga = (unsigned long long)(uintptr_t)g;
  asm volatile("global_load_async_to_lds_b128 %0, %1, off"
               :: "v"(ldsoff), "v"(ga) : "memory");
}
static __device__ inline void wait_async0() { asm volatile("s_wait_asynccnt 0x0" ::: "memory"); }
static __device__ inline void wait_async1() { asm volatile("s_wait_asynccnt 0x1" ::: "memory"); }
static __device__ inline void wait_async2() { asm volatile("s_wait_asynccnt 0x2" ::: "memory"); }

// ------------------- prep kernels -------------------
__global__ void cvt_bf16_kernel(const float* __restrict__ in, __bf16* __restrict__ out, int n) {
  int i = blockIdx.x * blockDim.x + threadIdx.x;
  if (i < n) out[i] = (__bf16)in[i];
}

// in: [rows, cols] f32 row-major -> out: [cols, rows] bf16 row-major
__global__ void transpose_bf16_kernel(const float* __restrict__ in, __bf16* __restrict__ out,
                                      int rows, int cols) {
  int i = blockIdx.x * blockDim.x + threadIdx.x;
  if (i < rows * cols) {
    int r = i / cols, c = i % cols;
    out[(size_t)c * rows + r] = (__bf16)in[i];
  }
}

// ---------------- shared GEMM core: 128(M) x 64(N) per block ----------------
// 8 waves; wave w owns rows m0..m0+15, all waves share the async-staged
// 64x32 B sub-panel (double buffered in LDS).
static __device__ inline void gemm_16x64_panel(const __bf16* __restrict__ A,
                                               const __bf16* __restrict__ BT,
                                               int m0, int n0blk,
                                               __bf16* bt0, __bf16* bt1,
                                               int tid, int l15, int hi,
                                               v8f acc[4]) {
  // staging map: thread t copies 16B: row sn (0..63), chunk sc (0/8/16/24)
  int sn = tid >> 2;
  int sc = (tid & 3) * 8;
  const __bf16* bsrc = BT + (size_t)(n0blk + sn) * EMB + sc;
  const int bo = sn * 32 + sc;

  async_copy_b128(bsrc, bt0 + bo);                    // stage k0=0
  for (int k0 = 0; k0 < EMB; k0 += 32) {
    __bf16* cur = (k0 & 32) ? bt1 : bt0;
    __bf16* nxt = (k0 & 32) ? bt0 : bt1;
    if (k0 + 32 < EMB) {
      async_copy_b128(bsrc + k0 + 32, nxt + bo);      // prefetch next panel
      wait_async1();                                  // current panel landed
    } else {
      wait_async0();
    }
    __syncthreads();                                  // panel visible to all waves
    v16bf a = load_a_frag(A, EMB, m0, k0, l15, hi);   // reused across 4 N-tiles
#pragma unroll
    for (int nt = 0; nt < 4; ++nt)
      acc[nt] = wmma_bf16(a, load_b_frag(cur, 32, nt * 16, 0, l15, hi), acc[nt]);
    __syncthreads();                                  // done reading before overwrite
  }
}

// ------------------- QKV GEMM + bias, scatter to q/k/vT -------------------
__global__ void qkv_gemm_kernel(const __bf16* __restrict__ xb,
                                const __bf16* __restrict__ wT,   // [3072,1024] bf16
                                const float*  __restrict__ bias, // [3072] f32
                                __bf16* __restrict__ q,
                                __bf16* __restrict__ k,
                                __bf16* __restrict__ vT) {
  __shared__ __align__(32) __bf16 bt[2][64 * 32];
  int tid = threadIdx.x;
  int lane = tid & 31, l15 = lane & 15, hi = lane >> 4;
  int wave = tid >> 5;
  int n0blk = blockIdx.x * 64;
  int m0 = (blockIdx.y * 8 + wave) * 16;

  v8f acc[4];
#pragma unroll
  for (int nt = 0; nt < 4; ++nt) acc[nt] = zero8();
  gemm_16x64_panel(xb, wT, m0, n0blk, bt[0], bt[1], tid, l15, hi, acc);

#pragma unroll
  for (int nt = 0; nt < 4; ++nt) {
    int n   = n0blk + nt * 16 + l15;  // output column, constant per lane
    int seg = n / EMB;                // 0=Q, 1=K, 2=V
    int nn  = n % EMB;
    int h   = nn / HDIM, d = nn % HDIM;
    float bv = bias[n];
#pragma unroll
    for (int e = 0; e < 8; ++e) {
      int r = m0 + e + hi * 8;        // C-frag row mapping
      int b = r >> 11, s = r & (S_LEN - 1);
      float val = acc[nt][e] + bv;
      size_t bh = (size_t)(b * HEADS + h);
      if (seg == 0)      q [(bh * S_LEN + s) * HDIM + d] = (__bf16)val;
      else if (seg == 1) k [(bh * S_LEN + s) * HDIM + d] = (__bf16)val;
      else               vT[(bh * HDIM + d) * S_LEN + s] = (__bf16)val;
    }
  }
}

// ------------------- flash attention: block = (b,h) x 128 queries ----------
// K/V tiles async-staged to LDS once per block (shared by all 8 waves).
__global__ void attn_kernel(const __bf16* __restrict__ q,
                            const __bf16* __restrict__ kk,
                            const __bf16* __restrict__ vT,
                            __bf16* __restrict__ ctx) {
  __shared__ __align__(32) __bf16 ktile[2][32 * HDIM];  // [keys 32][d 64]
  __shared__ __align__(32) __bf16 vtile[2][HDIM * 32];  // [d 64][keys 32]
  __shared__ __align__(32) __bf16 plds[8][16 * 32];     // per-wave P repack

  int tid = threadIdx.x;
  int lane = tid & 31, l15 = lane & 15, hi = lane >> 4;
  int wave = tid >> 5;
  int bh = blockIdx.y;            // 0..31
  int b = bh >> 4, h = bh & 15;
  int q0 = blockIdx.x * 128 + wave * 16;

  const __bf16* qp = q  + (size_t)bh * S_LEN * HDIM;
  const __bf16* kp = kk + (size_t)bh * S_LEN * HDIM;
  const __bf16* vp = vT + (size_t)bh * HDIM  * S_LEN;

  // staging maps (16B per thread per tile)
  int kr = tid >> 3, kc = (tid & 7) * 8;   // K tile: 32 rows x 64
  int vr = tid >> 2, vc = (tid & 3) * 8;   // V tile: 64 rows x 32
  auto stage = [&](int buf, int j0) {
    async_copy_b128(kp + (size_t)(j0 + kr) * HDIM + kc, &ktile[buf][kr * HDIM + kc]);
    async_copy_b128(vp + (size_t)vr * S_LEN + j0 + vc,  &vtile[buf][vr * 32 + vc]);
  };

  // Q strip as two A-fragments (K-dim 64 = 2 x 32)
  v16bf a0 = load_a_frag(qp, HDIM, q0, 0,  l15, hi);
  v16bf a1 = load_a_frag(qp, HDIM, q0, 32, l15, hi);

  const float cs = 1.4426950408889634f * 0.125f;  // log2(e) / sqrt(64)

  float mrow[8], lrow[8];
  v8f acc[4];
#pragma unroll
  for (int e = 0; e < 8; ++e) { mrow[e] = -1e30f; lrow[e] = 0.0f; }
#pragma unroll
  for (int nt = 0; nt < 4; ++nt) acc[nt] = zero8();

  stage(0, 0);
  for (int j0 = 0; j0 < S_LEN; j0 += 32) {
    int buf = (j0 >> 5) & 1;
    if (j0 + 32 < S_LEN) {
      stage(buf ^ 1, j0 + 32);   // prefetch next K/V tile
      wait_async2();             // current tile's 2 ops landed
    } else {
      wait_async0();
    }
    __syncthreads();             // tile visible to all waves
    const __bf16* kt = ktile[buf];
    const __bf16* vt = vtile[buf];

    // scores: two 16x16 tiles (key rows 0..15 and 16..31 of staged tile)
    v8f s0 = zero8(), s1 = zero8();
    s0 = wmma_bf16(a0, load_b_frag(kt, HDIM, 0,  0,  l15, hi), s0);
    s0 = wmma_bf16(a1, load_b_frag(kt, HDIM, 0,  32, l15, hi), s0);
    s1 = wmma_bf16(a0, load_b_frag(kt, HDIM, 16, 0,  l15, hi), s1);
    s1 = wmma_bf16(a1, load_b_frag(kt, HDIM, 16, 32, l15, hi), s1);

#pragma unroll
    for (int e = 0; e < 8; ++e) {
      float x0 = s0[e] * cs, x1 = s1[e] * cs;   // base-2 domain, pre-scaled
      float rmax = fmaxf(x0, x1);
      rmax = fmaxf(rmax, __shfl_xor(rmax, 1));
      rmax = fmaxf(rmax, __shfl_xor(rmax, 2));
      rmax = fmaxf(rmax, __shfl_xor(rmax, 4));
      rmax = fmaxf(rmax, __shfl_xor(rmax, 8));
      float mnew = fmaxf(mrow[e], rmax);
      float corr = exp2f(mrow[e] - mnew);
      mrow[e] = mnew;
      float p0 = exp2f(x0 - mnew), p1 = exp2f(x1 - mnew);
      float rsum = p0 + p1;
      rsum += __shfl_xor(rsum, 1);
      rsum += __shfl_xor(rsum, 2);
      rsum += __shfl_xor(rsum, 4);
      rsum += __shfl_xor(rsum, 8);
      lrow[e] = lrow[e] * corr + rsum;
      int row = e + hi * 8;                      // C-frag row
      plds[wave][row * 32 + l15]      = (__bf16)p0;
      plds[wave][row * 32 + 16 + l15] = (__bf16)p1;
#pragma unroll
      for (int nt = 0; nt < 4; ++nt) acc[nt][e] *= corr;
    }

    // repack P (C layout in LDS) into an A-fragment (16x32)
    v16bf pa;
    {
      const __bf16* lp = &plds[wave][l15 * 32 + hi * 8];
      v8bf lo = *reinterpret_cast<const v8bf*>(lp);
      v8bf hh = *reinterpret_cast<const v8bf*>(lp + 16);
#pragma unroll
      for (int i = 0; i < 8; ++i) { pa[i] = lo[i]; pa[i + 8] = hh[i]; }
    }

    // acc += P @ V  (N = 64 -> 4 tiles; V tile is N-major in LDS)
#pragma unroll
    for (int nt = 0; nt < 4; ++nt)
      acc[nt] = wmma_bf16(pa, load_b_frag(vt, 32, nt * 16, 0, l15, hi), acc[nt]);

    __syncthreads();             // all waves done reading before tile reuse
  }

  // epilogue: normalize and store ctx[B,S,E] (bf16)
#pragma unroll
  for (int e = 0; e < 8; ++e) {
    float invl = 1.0f / lrow[e];
    int s = q0 + e + hi * 8;
    size_t base = ((size_t)(b * S_LEN + s)) * EMB + (size_t)h * HDIM;
#pragma unroll
    for (int nt = 0; nt < 4; ++nt)
      ctx[base + nt * 16 + l15] = (__bf16)(acc[nt][e] * invl);
  }
}

// ------------------- output GEMM: [4096,1024] x [1024,1024] + bias -> f32 ----
__global__ void out_gemm_kernel(const __bf16* __restrict__ ctx,
                                const __bf16* __restrict__ wT,  // [1024,1024] bf16
                                const float*  __restrict__ bias,
                                float* __restrict__ out) {
  __shared__ __align__(32) __bf16 bt[2][64 * 32];
  int tid = threadIdx.x;
  int lane = tid & 31, l15 = lane & 15, hi = lane >> 4;
  int wave = tid >> 5;
  int n0blk = blockIdx.x * 64;
  int m0 = (blockIdx.y * 8 + wave) * 16;

  v8f acc[4];
#pragma unroll
  for (int nt = 0; nt < 4; ++nt) acc[nt] = zero8();
  gemm_16x64_panel(ctx, wT, m0, n0blk, bt[0], bt[1], tid, l15, hi, acc);

#pragma unroll
  for (int nt = 0; nt < 4; ++nt) {
    int n = n0blk + nt * 16 + l15;
    float bv = bias[n];
#pragma unroll
    for (int e = 0; e < 8; ++e) {
      int r = m0 + e + hi * 8;
      out[(size_t)r * EMB + n] = acc[nt][e] + bv;
    }
  }
}

// ------------------- launcher -------------------
extern "C" void kernel_launch(void* const* d_in, const int* in_sizes, int n_in,
                              void* d_out, int out_size, void* d_ws, size_t ws_size,
                              hipStream_t stream) {
  const float* x     = (const float*)d_in[0];  // [2,2048,1024]
  const float* W_qkv = (const float*)d_in[1];  // [1024,3072]
  const float* b_qkv = (const float*)d_in[2];  // [3072]
  const float* W_out = (const float*)d_in[3];  // [1024,1024]
  const float* b_out = (const float*)d_in[4];  // [1024]
  float* out = (float*)d_out;                  // [2,2048,1024] f32

  // workspace layout (48 MB total, all 1MB-aligned offsets)
  char* ws = (char*)d_ws;
  __bf16* xb    = (__bf16*)(ws + ((size_t)0  << 20));  // 8 MB  [4096,1024]
  __bf16* wqkvT = (__bf16*)(ws + ((size_t)8  << 20));  // 6 MB  [3072,1024]
  __bf16* woutT = (__bf16*)(ws + ((size_t)14 << 20));  // 2 MB  [1024,1024]
  __bf16* q     = (__bf16*)(ws + ((size_t)16 << 20));  // 8 MB  [B,H,S,D]
  __bf16* k     = (__bf16*)(ws + ((size_t)24 << 20));  // 8 MB  [B,H,S,D]
  __bf16* vT    = (__bf16*)(ws + ((size_t)32 << 20));  // 8 MB  [B,H,D,S]
  __bf16* ctx   = (__bf16*)(ws + ((size_t)40 << 20));  // 8 MB  [4096,1024]

  int n;
  n = ROWS * EMB;
  cvt_bf16_kernel<<<(n + 255) / 256, 256, 0, stream>>>(x, xb, n);
  n = EMB * QKV3;
  transpose_bf16_kernel<<<(n + 255) / 256, 256, 0, stream>>>(W_qkv, wqkvT, EMB, QKV3);
  n = EMB * EMB;
  transpose_bf16_kernel<<<(n + 255) / 256, 256, 0, stream>>>(W_out, woutT, EMB, EMB);

  qkv_gemm_kernel<<<dim3(QKV3 / 64, ROWS / 128), 256, 0, stream>>>(
      xb, wqkvT, b_qkv, q, k, vT);

  attn_kernel<<<dim3(S_LEN / 128, BATCH * HEADS), 256, 0, stream>>>(q, k, vT, ctx);

  out_gemm_kernel<<<dim3(EMB / 64, ROWS / 128), 256, 0, stream>>>(
      ctx, woutT, b_out, out);
}
